// CustomAttention_58437325029385
// MI455X (gfx1250) — matrix operand order repaired and checked by reading, not compiled, not run on hardware
//
#include <hip/hip_runtime.h>
#include <hip/hip_bf16.h>

// ---------------------------------------------------------------------------
// Flash attention for B=4, H=16, N=2048, D=64 on gfx1250 (MI455X).
// Transposed formulation: S^T = K*Q^T, O^T = V^T*P^T; each lane owns one
// query column per query-group.  TWO query tiles per wave (BLK_M=256) so
// every LDS tile / operand load feeds 2x the WMMA work and K/V global
// re-reads are halved.  bf16 WMMA 16x16x32, log2-domain online softmax,
// double-buffered LDS tiles, WGP-scope prefetch of the next tiles.
// ---------------------------------------------------------------------------

typedef __attribute__((ext_vector_type(16))) __bf16          v16bf;
typedef __attribute__((ext_vector_type(2)))  __bf16          v2bf;
typedef __attribute__((ext_vector_type(2)))  float           v2f;
typedef __attribute__((ext_vector_type(8)))  float           v8f;
typedef __attribute__((ext_vector_type(16))) unsigned short  v16u16;
typedef __attribute__((ext_vector_type(8)))  unsigned short  v8u16;
typedef __attribute__((ext_vector_type(8)))  unsigned int    v8u32;

namespace {
constexpr int Nn    = 2048;
constexpr int Hh    = 16;
constexpr int Dd    = 64;
constexpr int HD    = Hh * Dd;    // 1024
constexpr int BLK_M = 256;        // q rows per workgroup (8 waves * 2 * 16)
constexpr int QOFF  = 128;        // offset between a wave's two query tiles
constexpr int BLK_N = 32;         // keys per inner iteration
// 1/sqrt(64) * log2(e): softmax runs in the exp2 domain.
constexpr float SCALE_L2E = 0.125f * 1.44269504088896340736f;
}

#if defined(__has_builtin)
#if __has_builtin(__builtin_amdgcn_cvt_pk_bf16_f32)
#define HAVE_CVT_PK_BF16 1
#endif
#if __has_builtin(__builtin_amdgcn_exp2f)
#define HAVE_EXP2 1
#endif
#endif

__device__ __forceinline__ unsigned int pk2bf(float lo, float hi) {
#ifdef HAVE_CVT_PK_BF16
  auto r = __builtin_amdgcn_cvt_pk_bf16_f32(lo, hi);   // v_cvt_pk_bf16_f32
  return __builtin_bit_cast(unsigned int, r);
#else
  v2f t = {lo, hi};
  v2bf r = __builtin_convertvector(t, v2bf);           // packed fptrunc
  return __builtin_bit_cast(unsigned int, r);
#endif
}

__device__ __forceinline__ float fast_exp2(float x) {
#ifdef HAVE_EXP2
  return __builtin_amdgcn_exp2f(x);                    // raw v_exp_f32
#else
  return __builtin_exp2f(x);
#endif
}

__device__ __forceinline__ v16bf as_bf(v16u16 x) {
  return __builtin_bit_cast(v16bf, x);
}
__device__ __forceinline__ v16bf as_bf32(v8u32 x) {
  return __builtin_bit_cast(v16bf, x);
}
__device__ __forceinline__ v16bf cat8(v8u16 lo, v8u16 hi) {
  return as_bf((v16u16)__builtin_shufflevector(
      lo, hi, 0,1,2,3,4,5,6,7,8,9,10,11,12,13,14,15));
}

// Online-softmax update for one query group; returns the P^T B-operand.
__device__ __forceinline__ v16bf softmax_update(
    const v8f& s0, const v8f& s1, float& rmax, float& rsum,
    v8f& o0, v8f& o1, v8f& o2, v8f& o3) {
  float m = fmaxf(s0[0], s0[1]);
#pragma unroll
  for (int j = 2; j < 8; ++j) m = fmaxf(m, s0[j]);
#pragma unroll
  for (int j = 0; j < 8; ++j) m = fmaxf(m, s1[j]);
  m = fmaxf(m, __shfl_xor(m, 16));          // partner lane: other 16 keys

  const float nm = fmaxf(rmax, m);
  const float al = fast_exp2(rmax - nm);
  rmax = nm;

  float p0[8], p1[8], ssum = 0.0f;
#pragma unroll
  for (int j = 0; j < 8; ++j) {
    p0[j] = fast_exp2(s0[j] - nm);
    p1[j] = fast_exp2(s1[j] - nm);
    ssum += p0[j] + p1[j];
  }
  ssum += __shfl_xor(ssum, 16);
  rsum = rsum * al + ssum;
#pragma unroll
  for (int j = 0; j < 8; ++j) {
    o0[j] *= al; o1[j] *= al; o2[j] *= al; o3[j] *= al;
  }

  v8u32 pdw;
#pragma unroll
  for (int j = 0; j < 4; ++j) {
    pdw[j]     = pk2bf(p0[2 * j], p0[2 * j + 1]);
    pdw[j + 4] = pk2bf(p1[2 * j], p1[2 * j + 1]);
  }
  return as_bf32(pdw);
}

__device__ __forceinline__ void write_out(
    unsigned short* __restrict__ Op, size_t row, int dimoff, float rsum,
    const v8f& o0, const v8f& o1, const v8f& o2, const v8f& o3) {
  const float inv = 1.0f / rsum;
  const size_t ro = row * HD + dimoff;
  uint4 w;
  w.x = pk2bf(o0[0] * inv, o0[1] * inv);
  w.y = pk2bf(o0[2] * inv, o0[3] * inv);
  w.z = pk2bf(o0[4] * inv, o0[5] * inv);
  w.w = pk2bf(o0[6] * inv, o0[7] * inv);
  *(uint4*)&Op[ro] = w;
  w.x = pk2bf(o1[0] * inv, o1[1] * inv);
  w.y = pk2bf(o1[2] * inv, o1[3] * inv);
  w.z = pk2bf(o1[4] * inv, o1[5] * inv);
  w.w = pk2bf(o1[6] * inv, o1[7] * inv);
  *(uint4*)&Op[ro + 16] = w;
  w.x = pk2bf(o2[0] * inv, o2[1] * inv);
  w.y = pk2bf(o2[2] * inv, o2[3] * inv);
  w.z = pk2bf(o2[4] * inv, o2[5] * inv);
  w.w = pk2bf(o2[6] * inv, o2[7] * inv);
  *(uint4*)&Op[ro + 32] = w;
  w.x = pk2bf(o3[0] * inv, o3[1] * inv);
  w.y = pk2bf(o3[2] * inv, o3[3] * inv);
  w.z = pk2bf(o3[4] * inv, o3[5] * inv);
  w.w = pk2bf(o3[6] * inv, o3[7] * inv);
  *(uint4*)&Op[ro + 48] = w;
}

__global__ __launch_bounds__(256)
void flash_attn_bf16_wmma(const float* __restrict__ qg,
                          const float* __restrict__ kg,
                          const float* __restrict__ vg,
                          unsigned short* __restrict__ outg) {
  // Double-buffered tiles: K [key][d] bf16, V transposed [d][key] bf16.
  __shared__ __align__(32) unsigned short kt[2][BLK_N * Dd];   // 2 * 4KB
  __shared__ __align__(32) unsigned short vt[2][Dd * BLK_N];   // 2 * 4KB

  const int tid  = threadIdx.x;
  const int wv   = tid >> 5;
  const int lane = tid & 31;
  const int half = lane >> 4;          // 0: lanes 0-15, 1: lanes 16-31
  const int l16  = lane & 15;

  const int b = blockIdx.z;
  const int h = blockIdx.y;
  const int q0 = blockIdx.x * BLK_M + wv * 16;   // query tile A; tile B at +QOFF

  const size_t bh = (size_t)b * Nn * HD + (size_t)h * Dd;
  const float* Qp = qg + bh;
  const float* Kp = kg + bh;
  const float* Vp = vg + bh;
  unsigned short* Op = outg + bh;

  // ---- Q^T B-operands for both query tiles (pre-scaled, bf16) ------------
  const int db = half ? 16 : 0;
  const float* qrA = Qp + (size_t)(q0 + l16) * HD;
  const float* qrB = qrA + (size_t)QOFF * HD;
  v8u32 a0, a1, b0, b1;
#pragma unroll
  for (int j = 0; j < 8; ++j) {
    a0[j] = pk2bf(qrA[db + 2 * j] * SCALE_L2E,      qrA[db + 2 * j + 1] * SCALE_L2E);
    a1[j] = pk2bf(qrA[32 + db + 2 * j] * SCALE_L2E, qrA[32 + db + 2 * j + 1] * SCALE_L2E);
    b0[j] = pk2bf(qrB[db + 2 * j] * SCALE_L2E,      qrB[db + 2 * j + 1] * SCALE_L2E);
    b1[j] = pk2bf(qrB[32 + db + 2 * j] * SCALE_L2E, qrB[32 + db + 2 * j + 1] * SCALE_L2E);
  }
  const v16bf bqA0 = as_bf32(a0), bqA1 = as_bf32(a1);
  const v16bf bqB0 = as_bf32(b0), bqB1 = as_bf32(b1);

  // ---- per-lane online-softmax state: one query per lane per group -------
  v8f z = {0,0,0,0,0,0,0,0};
  v8f o0A = z, o1A = z, o2A = z, o3A = z;
  v8f o0B = z, o1B = z, o2B = z, o3B = z;
  float rmaxA = -3.0e38f, rsumA = 0.0f;
  float rmaxB = -3.0e38f, rsumB = 0.0f;

  // cooperative loader: each thread owns a 2-key x 4-dim block.
  const int dblk  = tid & 15;
  const int kpair = tid >> 4;
  const int ck0   = 2 * kpair;
  const int cd0   = 4 * dblk;
  // key column for V^T: unpermuted key order
  const int cgrp  = ck0 >> 3;
  const int pgrp  = cgrp;
  const int pk0   = pgrp * 8 + (ck0 & 7);

  const int cb = half ? 8 : 0;     // A-operand column-group select

  constexpr int NITER = Nn / BLK_N;  // 64

  auto fill = [&](int buf, int it) {
    const float* kr0p = Kp + (size_t)(it * BLK_N + ck0) * HD + cd0;
    const float* kr1p = kr0p + HD;
    const float4 ka  = *(const float4*)kr0p;
    const float4 kb2 = *(const float4*)kr1p;
    uint2 pka, pkb;
    pka.x = pk2bf(ka.x, ka.y);
    pka.y = pk2bf(ka.z, ka.w);
    pkb.x = pk2bf(kb2.x, kb2.y);
    pkb.y = pk2bf(kb2.z, kb2.w);
    *(uint2*)&kt[buf][ck0 * Dd + cd0]       = pka;
    *(uint2*)&kt[buf][(ck0 + 1) * Dd + cd0] = pkb;

    const float* vr0p = Vp + (size_t)(it * BLK_N + ck0) * HD + cd0;
    const float* vr1p = vr0p + HD;
    const float4 va  = *(const float4*)vr0p;
    const float4 vb2 = *(const float4*)vr1p;
    unsigned short* vrow = &vt[buf][pk0];
    *(unsigned int*)(vrow + (cd0 + 0) * BLK_N) = pk2bf(va.x, vb2.x);
    *(unsigned int*)(vrow + (cd0 + 1) * BLK_N) = pk2bf(va.y, vb2.y);
    *(unsigned int*)(vrow + (cd0 + 2) * BLK_N) = pk2bf(va.z, vb2.z);
    *(unsigned int*)(vrow + (cd0 + 3) * BLK_N) = pk2bf(va.w, vb2.w);

    if (it + 1 < NITER) {
      // WGP-scope (near-cache) prefetch of the tile after next, so the
      // following global_load_b128s hit close to the WGP.
      __builtin_prefetch(kr0p + (size_t)BLK_N * HD, 0, 3);
      __builtin_prefetch(vr0p + (size_t)BLK_N * HD, 0, 3);
    }
  };

  fill(0, 0);
  __syncthreads();

  for (int it = 0; it < NITER; ++it) {
    const int cur = it & 1;
    const unsigned short* ktc = kt[cur];
    const unsigned short* vtc = vt[cur];

    // ---- S^T tiles for both query groups: shared K A-operands ------------
    v8f s0A = z, s1A = z, s0B = z, s1B = z;
    {
      const unsigned short* kr0 = &ktc[l16 * Dd];
      const unsigned short* kr1 = &ktc[(16 + l16) * Dd];
      const v16bf ak00 = cat8(*(const v8u16*)&kr0[2 * cb],      *(const v8u16*)&kr0[2 * cb + 8]);
      const v16bf ak01 = cat8(*(const v8u16*)&kr0[32 + 2 * cb], *(const v8u16*)&kr0[40 + 2 * cb]);
      const v16bf ak10 = cat8(*(const v8u16*)&kr1[2 * cb],      *(const v8u16*)&kr1[2 * cb + 8]);
      const v16bf ak11 = cat8(*(const v8u16*)&kr1[32 + 2 * cb], *(const v8u16*)&kr1[40 + 2 * cb]);
      s0A = __builtin_amdgcn_wmma_f32_16x16x32_bf16(false, ak00, false, bqA0,
                                                    (short)0, s0A, false, false);
      s0B = __builtin_amdgcn_wmma_f32_16x16x32_bf16(false, ak00, false, bqB0,
                                                    (short)0, s0B, false, false);
      s0A = __builtin_amdgcn_wmma_f32_16x16x32_bf16(false, ak01, false, bqA1,
                                                    (short)0, s0A, false, false);
      s0B = __builtin_amdgcn_wmma_f32_16x16x32_bf16(false, ak01, false, bqB1,
                                                    (short)0, s0B, false, false);
      s1A = __builtin_amdgcn_wmma_f32_16x16x32_bf16(false, ak10, false, bqA0,
                                                    (short)0, s1A, false, false);
      s1B = __builtin_amdgcn_wmma_f32_16x16x32_bf16(false, ak10, false, bqB0,
                                                    (short)0, s1B, false, false);
      s1A = __builtin_amdgcn_wmma_f32_16x16x32_bf16(false, ak11, false, bqA1,
                                                    (short)0, s1A, false, false);
      s1B = __builtin_amdgcn_wmma_f32_16x16x32_bf16(false, ak11, false, bqB1,
                                                    (short)0, s1B, false, false);
    }

    // ---- softmax + P^T operands (in-register, permuted key order) --------
    const v16bf bpA = softmax_update(s0A, s1A, rmaxA, rsumA, o0A, o1A, o2A, o3A);
    const v16bf bpB = softmax_update(s0B, s1B, rmaxB, rsumB, o0B, o1B, o2B, o3B);

    // ---- O^T += V^T * P^T : shared V A-operands, both query groups -------
    {
      const unsigned short* v0r = &vtc[(l16)      * BLK_N];
      const unsigned short* v1r = &vtc[(16 + l16) * BLK_N];
      const unsigned short* v2r = &vtc[(32 + l16) * BLK_N];
      const unsigned short* v3r = &vtc[(48 + l16) * BLK_N];
      const v16bf av0 = cat8(*(const v8u16*)&v0r[cb], *(const v8u16*)&v0r[cb + 16]);
      const v16bf av1 = cat8(*(const v8u16*)&v1r[cb], *(const v8u16*)&v1r[cb + 16]);
      const v16bf av2 = cat8(*(const v8u16*)&v2r[cb], *(const v8u16*)&v2r[cb + 16]);
      const v16bf av3 = cat8(*(const v8u16*)&v3r[cb], *(const v8u16*)&v3r[cb + 16]);
      o0A = __builtin_amdgcn_wmma_f32_16x16x32_bf16(false, av0, false, bpA,
                                                    (short)0, o0A, false, false);
      o0B = __builtin_amdgcn_wmma_f32_16x16x32_bf16(false, av0, false, bpB,
                                                    (short)0, o0B, false, false);
      o1A = __builtin_amdgcn_wmma_f32_16x16x32_bf16(false, av1, false, bpA,
                                                    (short)0, o1A, false, false);
      o1B = __builtin_amdgcn_wmma_f32_16x16x32_bf16(false, av1, false, bpB,
                                                    (short)0, o1B, false, false);
      o2A = __builtin_amdgcn_wmma_f32_16x16x32_bf16(false, av2, false, bpA,
                                                    (short)0, o2A, false, false);
      o2B = __builtin_amdgcn_wmma_f32_16x16x32_bf16(false, av2, false, bpB,
                                                    (short)0, o2B, false, false);
      o3A = __builtin_amdgcn_wmma_f32_16x16x32_bf16(false, av3, false, bpA,
                                                    (short)0, o3A, false, false);
      o3B = __builtin_amdgcn_wmma_f32_16x16x32_bf16(false, av3, false, bpB,
                                                    (short)0, o3B, false, false);
    }

    if (it + 1 < NITER) fill((it + 1) & 1, it + 1);
    __syncthreads();
  }

  // ---- epilogue: both query groups ---------------------------------------
  const int dimoff = 8 * half;
  write_out(Op, (size_t)(q0 + l16),        dimoff, rsumA, o0A, o1A, o2A, o3A);
  write_out(Op, (size_t)(q0 + QOFF + l16), dimoff, rsumB, o0B, o1B, o2B, o3B);
}

extern "C" void kernel_launch(void* const* d_in, const int* in_sizes, int n_in,
                              void* d_out, int out_size, void* d_ws, size_t ws_size,
                              hipStream_t stream) {
  (void)in_sizes; (void)n_in; (void)out_size; (void)d_ws; (void)ws_size;
  const float* q = (const float*)d_in[0];
  const float* k = (const float*)d_in[1];
  const float* v = (const float*)d_in[2];
  unsigned short* out = (unsigned short*)d_out;   // bf16 payload

  dim3 grid(Nn / BLK_M, Hh, 4);   // (8, 16, 4) = 512 workgroups
  dim3 block(256);                // 8 wave32
  flash_attn_bf16_wmma<<<grid, block, 0, stream>>>(q, k, v, out);
}
